// LAINRDecoder_53085795779215
// MI455X (gfx1250) — matrix operand order repaired and checked by reading, not compiled, not run on hardware
//
#include <hip/hip_runtime.h>
#include <hip/hip_bf16.h>

typedef __attribute__((ext_vector_type(16))) _Float16 v16h;
typedef __attribute__((ext_vector_type(8)))  _Float16 v8h;
typedef __attribute__((ext_vector_type(4)))  _Float16 v4h;
typedef __attribute__((ext_vector_type(8)))  float    v8f;
typedef __attribute__((ext_vector_type(4)))  int      v4i;

#define PI_F 3.14159265358979323846f

constexpr int BATCH = 2;
constexpr int NQ    = 4096;
constexpr int NL    = 1024;
constexpr int HDIM  = 256;   // hidden dim
constexpr int FDIM  = 64;    // fourier feature dim
constexpr int NHEAD = 2;
constexpr int DHEAD = 64;
constexpr int INNER = NHEAD * DHEAD;   // 128
constexpr int KVW   = 2 * INNER;       // 256 (k cols 0..127, v cols 128..255)
constexpr int TOPK  = 128;

// gfx1250 async global->LDS path (ASYNCcnt); guarded so compile never breaks.
#if __has_builtin(__builtin_amdgcn_global_load_async_to_lds_b128) && \
    __has_builtin(__builtin_amdgcn_s_wait_asynccnt)
#define ASYNC_LDS 1
typedef __attribute__((address_space(1))) v4i* as1_v4i_ptr;
typedef __attribute__((address_space(3))) v4i* as3_v4i_ptr;
#else
#define ASYNC_LDS 0
#endif

// ---------------------------------------------------------------------------
// Per-query precompute: window start (closed-form replacement for top_k) and
// Fourier features for the two omega scales (128 and 32), stored as f16.
// ---------------------------------------------------------------------------
__global__ void k_precompute(const float* __restrict__ x,
                             const int* gDp, const int* gHp,
                             const int* gWp, const int* gTp,
                             _Float16* __restrict__ g128,
                             _Float16* __restrict__ g32,
                             int* __restrict__ win) {
    int q = blockIdx.x * blockDim.x + threadIdx.x;
    if (q >= NQ) return;
    float g0 = x[q * 4 + 0], g1 = x[q * 4 + 1];
    float g2 = x[q * 4 + 2], g3 = x[q * 4 + 3];
    int gD = *gDp, gH = *gHp, gW = *gWp, gT = *gTp;
    int zi = (int)(g0 * gD), yi = (int)(g1 * gH);
    int xi = (int)(g2 * gW), ti = (int)(g3 * gT);
    int idx = ((ti * gD + zi) * gH + yi) * gW + xi;
    int Ntot = gD * gH * gW * gT;
    // 128 token positions closest to u -> contiguous window, start = ceil(u-64)
    float u = (float)idx * ((float)NL / (float)Ntot) - 0.5f;
    int s = (int)ceilf(u - (float)(TOPK / 2));
    s = s < 0 ? 0 : s;
    s = s > (NL - TOPK) ? (NL - TOPK) : s;
    win[q] = s;

    const float LN10 = 2.302585092994046f;
    float bexp[2];
    bexp[0] = (log10f(128.0f) - 1.0f) / 7.0f;
    bexp[1] = (log10f(32.0f)  - 1.0f) / 7.0f;
    _Float16* outs[2] = { g128 + (size_t)q * FDIM, g32 + (size_t)q * FDIM };
    float gs[4] = { g0, g1, g2, g3 };
    for (int set = 0; set < 2; ++set) {
        for (int f = 0; f < 8; ++f) {
            float w = __expf((1.0f + (float)f * bexp[set]) * LN10);
            for (int d = 0; d < 4; ++d) {
                float a = PI_F * gs[d] * w;
                outs[set][d * 16 + f]     = (_Float16)sinf(a);
                outs[set][d * 16 + 8 + f] = (_Float16)cosf(a);
            }
        }
    }
}

// ---------------------------------------------------------------------------
__global__ void k_f32_to_f16(const float* __restrict__ in,
                             _Float16* __restrict__ out, int n) {
    int i = blockIdx.x * blockDim.x + threadIdx.x;
    if (i < n) out[i] = (_Float16)in[i];
}

// Transpose-convert weight W[K][N] (f32, row major) -> Wt[N][K] (f16)
__global__ void k_transpose_f16(const float* __restrict__ W,
                                _Float16* __restrict__ Wt, int K, int N) {
    int i = blockIdx.x * blockDim.x + threadIdx.x;
    if (i >= K * N) return;
    int n = i / K, k = i % K;
    Wt[i] = (_Float16)W[k * N + n];
}

__global__ void k_sum_f16(const float* __restrict__ a,
                          const float* __restrict__ b,
                          _Float16* __restrict__ o, int n) {
    int i = blockIdx.x * blockDim.x + threadIdx.x;
    if (i < n) o[i] = (_Float16)(a[i] + b[i]);
}

// ---------------------------------------------------------------------------
// WMMA GEMM: out = act( A(MxK,f16) * Wt(NxK,f16)^T + bias + addend )
// Block = 64 rows x 128 cols, 8 waves. The 64xK A-tile is staged once into
// LDS (async global->LDS when available); each wave owns one 16-col strip and
// 4 M-subtiles, so each global B-fragment load feeds 4 WMMAs. All 8 A-fragment
// LDS loads are issued before the 4 WMMAs so the scheduler can clause them and
// overlap DScnt waits with the first WMMAs.
// ---------------------------------------------------------------------------
__global__ __launch_bounds__(256) void k_gemm_wmma(
    const _Float16* __restrict__ A, const _Float16* __restrict__ Wt,
    const float* __restrict__ bias, const float* __restrict__ addend,
    int addRowMod, float* __restrict__ outF, _Float16* __restrict__ outH,
    int M, int N, int K, int doRelu) {
    __shared__ _Float16 ldsA[64 * 256];          // 32 KB max (K <= 256)
    const int tid  = threadIdx.x;
    const int lane = threadIdx.x & 31;
    const int wave = threadIdx.x >> 5;
    const int m0 = blockIdx.x * 64;
    const int n0 = blockIdx.y * 128 + wave * 16;

    // ---- stage A tile (64 x K halves) into LDS ----
    const int k8 = K >> 3;                       // b128 chunks per row
    const int chunks = 64 * k8;                  // multiple of 256 for all K
    for (int c = tid; c < chunks; c += 256) {
        int row = c / k8;
        int col = (c - row * k8) * 8;
        const _Float16* src = A + (size_t)(m0 + row) * K + col;
        _Float16* dst = &ldsA[row * K + col];
#if ASYNC_LDS
        __builtin_amdgcn_global_load_async_to_lds_b128(
            (as1_v4i_ptr)src, (as3_v4i_ptr)dst, 0, 0);
#else
        *(v8h*)dst = *(const v8h*)src;
#endif
    }
#if ASYNC_LDS
    __builtin_amdgcn_s_wait_asynccnt(0);
#endif
    __syncthreads();
    if (n0 >= N) return;

    const int hh  = lane >> 4;   // half-wave selector
    const int l15 = lane & 15;

    // A fragment rows (from LDS): lane -> M = l15 (+16*t); K chunk 16*hh..16*hh+15
    const _Float16* a0 = &ldsA[(size_t)l15 * K + 16 * hh];
    // B fragment: lane -> N = l15; K chunk 16*hh..16*hh+15 (Wt is N-major)
    const _Float16* brow = Wt + (size_t)(n0 + l15) * K + 16 * hh;

    v8f c0 = {}, c1 = {}, c2 = {}, c3 = {};
    for (int kk = 0; kk < K; kk += 32) {
        // issue every load for this K-step first, WMMAs after
        v16h bfrag = *(const v16h*)(brow + kk);
        v8h lo0 = *(const v8h*)(a0 + kk);
        v8h hi0 = *(const v8h*)(a0 + kk + 8);
        v8h lo1 = *(const v8h*)(a0 + 16 * K + kk);
        v8h hi1 = *(const v8h*)(a0 + 16 * K + kk + 8);
        v8h lo2 = *(const v8h*)(a0 + 32 * K + kk);
        v8h hi2 = *(const v8h*)(a0 + 32 * K + kk + 8);
        v8h lo3 = *(const v8h*)(a0 + 48 * K + kk);
        v8h hi3 = *(const v8h*)(a0 + 48 * K + kk + 8);
        v16h af0 = __builtin_shufflevector(lo0, hi0,
                       0,1,2,3,4,5,6,7,8,9,10,11,12,13,14,15);
        v16h af1 = __builtin_shufflevector(lo1, hi1,
                       0,1,2,3,4,5,6,7,8,9,10,11,12,13,14,15);
        v16h af2 = __builtin_shufflevector(lo2, hi2,
                       0,1,2,3,4,5,6,7,8,9,10,11,12,13,14,15);
        v16h af3 = __builtin_shufflevector(lo3, hi3,
                       0,1,2,3,4,5,6,7,8,9,10,11,12,13,14,15);
        c0 = __builtin_amdgcn_wmma_f32_16x16x32_f16(
                 false, af0, false, bfrag, (short)0, c0, false, false);
        c1 = __builtin_amdgcn_wmma_f32_16x16x32_f16(
                 false, af1, false, bfrag, (short)0, c1, false, false);
        c2 = __builtin_amdgcn_wmma_f32_16x16x32_f16(
                 false, af2, false, bfrag, (short)0, c2, false, false);
        c3 = __builtin_amdgcn_wmma_f32_16x16x32_f16(
                 false, af3, false, bfrag, (short)0, c3, false, false);
    }

    const int nn = n0 + l15;
    const float bv = bias ? bias[nn] : 0.0f;
    v8f accs[4] = { c0, c1, c2, c3 };
    for (int t = 0; t < 4; ++t) {
        for (int r = 0; r < 8; ++r) {
            int mm = m0 + t * 16 + r + 8 * hh;   // C/D layout: VGPR r, +8 per half
            float v = accs[t][r] + bv;
            if (addend) {
                int ar = addRowMod ? (mm % addRowMod) : mm;
                v += addend[(size_t)ar * N + nn];
            }
            if (doRelu) v = fmaxf(v, 0.0f);
            size_t o = (size_t)mm * N + nn;
            if (outF) outF[o] = v;
            if (outH) outH[o] = (_Float16)v;
        }
    }
}

// ---------------------------------------------------------------------------
// Windowed attention: one wave per (b,q). Keys/values are the contiguous
// 128-token window starting at win[q]; kv16 rows hold [k(0..127) v(128..255)].
// ---------------------------------------------------------------------------
__global__ __launch_bounds__(256) void k_attn(
    const float* __restrict__ qv, const _Float16* __restrict__ kv16,
    const int* __restrict__ win, _Float16* __restrict__ obuf) {
    __shared__ float lds_q[8][INNER];
    __shared__ float lds_w[8][2][TOPK];
    const int lane = threadIdx.x & 31;
    const int wave = threadIdx.x >> 5;
    const int row  = blockIdx.x * 8 + wave;   // row = b*NQ + q
    const int b    = row / NQ;
    const int qq   = row % NQ;

    float4 qvec = *(const float4*)(qv + (size_t)qq * INNER + 4 * lane);
    *(float4*)(&lds_q[wave][4 * lane]) = qvec;
    __syncthreads();

    const int s = win[qq];
    const _Float16* kvb = kv16 + (size_t)(b * NL + s) * KVW;

    // pass 1: sims + softmax (4 keys per lane)
    float sim0[4], sim1[4];
    for (int j = 0; j < 4; ++j) {
        const int key = lane + 32 * j;
        const _Float16* krow = kvb + (size_t)key * KVW;
        if (j < 3)
            __builtin_prefetch((const void*)(kvb + (size_t)(key + 32) * KVW), 0, 1);
        float a0 = 0.0f, a1 = 0.0f;
        for (int d8 = 0; d8 < 8; ++d8) {         // head 0: dims 0..63
            v8h kh = *(const v8h*)(krow + 8 * d8);
#pragma unroll
            for (int i = 0; i < 8; ++i) a0 += (float)kh[i] * lds_q[wave][8 * d8 + i];
        }
        for (int d8 = 8; d8 < 16; ++d8) {        // head 1: dims 64..127
            v8h kh = *(const v8h*)(krow + 8 * d8);
#pragma unroll
            for (int i = 0; i < 8; ++i) a1 += (float)kh[i] * lds_q[wave][8 * d8 + i];
        }
        sim0[j] = a0 * 0.125f;   // DH^-0.5
        sim1[j] = a1 * 0.125f;
    }
    float mx0 = fmaxf(fmaxf(sim0[0], sim0[1]), fmaxf(sim0[2], sim0[3]));
    float mx1 = fmaxf(fmaxf(sim1[0], sim1[1]), fmaxf(sim1[2], sim1[3]));
    for (int off = 16; off; off >>= 1) {
        mx0 = fmaxf(mx0, __shfl_xor(mx0, off, 32));
        mx1 = fmaxf(mx1, __shfl_xor(mx1, off, 32));
    }
    float e0[4], e1[4], sum0 = 0.0f, sum1 = 0.0f;
    for (int j = 0; j < 4; ++j) {
        e0[j] = __expf(sim0[j] - mx0); sum0 += e0[j];
        e1[j] = __expf(sim1[j] - mx1); sum1 += e1[j];
    }
    for (int off = 16; off; off >>= 1) {
        sum0 += __shfl_xor(sum0, off, 32);
        sum1 += __shfl_xor(sum1, off, 32);
    }
    float inv0 = 1.0f / sum0, inv1 = 1.0f / sum1;
    for (int j = 0; j < 4; ++j) {
        lds_w[wave][0][lane + 32 * j] = e0[j] * inv0;
        lds_w[wave][1][lane + 32 * j] = e1[j] * inv1;
    }
    __syncthreads();

    // pass 2: o[d] = sum_k w[h(d)][k] * v[k][d]; lane handles dims 4*lane..+3
    const int d0 = 4 * lane;
    const int head = lane >> 4;          // d0/64
    float acc0 = 0.f, acc1 = 0.f, acc2 = 0.f, acc3 = 0.f;
    const _Float16* vb = kvb + INNER + d0;
    for (int k = 0; k < TOPK; ++k) {
        float w = lds_w[wave][head][k];
        v4h vv = *(const v4h*)(vb + (size_t)k * KVW);
        acc0 += w * (float)vv[0];
        acc1 += w * (float)vv[1];
        acc2 += w * (float)vv[2];
        acc3 += w * (float)vv[3];
    }
    v4h o;
    o[0] = (_Float16)acc0; o[1] = (_Float16)acc1;
    o[2] = (_Float16)acc2; o[3] = (_Float16)acc3;
    *(v4h*)(obuf + (size_t)row * INNER + d0) = o;
}

// ---------------------------------------------------------------------------
__global__ __launch_bounds__(256) void k_final(
    const float* __restrict__ m0, const float* __restrict__ hv1,
    const float* __restrict__ outlW, const float* __restrict__ outlb,
    float* __restrict__ out) {
    const int lane = threadIdx.x & 31;
    const int wave = threadIdx.x >> 5;
    const int row  = blockIdx.x * 8 + wave;
    float acc = 0.0f;
    for (int j = 0; j < 8; ++j) {
        int i = lane + 32 * j;
        acc += m0[(size_t)row * HDIM + i]  * outlW[i];
        acc += hv1[(size_t)row * HDIM + i] * outlW[HDIM + i];
    }
    for (int off = 16; off; off >>= 1) acc += __shfl_xor(acc, off, 32);
    if (lane == 0) out[row] = acc + outlb[0] + outlb[1];
}

// ---------------------------------------------------------------------------
extern "C" void kernel_launch(void* const* d_in, const int* in_sizes, int n_in,
                              void* d_out, int out_size, void* d_ws, size_t ws_size,
                              hipStream_t stream) {
    (void)in_sizes; (void)n_in; (void)out_size; (void)ws_size;
    const float* x       = (const float*)d_in[0];
    const float* tokens  = (const float*)d_in[1];
    const float* query_W = (const float*)d_in[2];
    const float* query_b = (const float*)d_in[3];
    const float* q_W     = (const float*)d_in[4];
    const float* kv_W    = (const float*)d_in[5];
    const float* out_W   = (const float*)d_in[6];
    const float* out_b   = (const float*)d_in[7];
    const float* band_W  = (const float*)d_in[8];
    const float* band_b  = (const float*)d_in[9];
    const float* mod_W   = (const float*)d_in[10];
    const float* mod_b   = (const float*)d_in[11];
    const float* hv_W    = (const float*)d_in[12];
    const float* hv_b    = (const float*)d_in[13];
    const float* outl_W  = (const float*)d_in[14];
    const float* outl_b  = (const float*)d_in[15];
    const int* gD = (const int*)d_in[16];
    const int* gH = (const int*)d_in[17];
    const int* gW = (const int*)d_in[18];
    const int* gT = (const int*)d_in[19];
    float* outp = (float*)d_out;

    size_t off = 0;
    auto alloc = [&](size_t bytes) -> void* {
        off = (off + 255) & ~(size_t)255;
        void* p = (char*)d_ws + off;
        off += bytes;
        return p;
    };
    const int BQ = BATCH * NQ;
    _Float16* g128  = (_Float16*)alloc((size_t)NQ * FDIM * 2);
    _Float16* g32   = (_Float16*)alloc((size_t)NQ * FDIM * 2);
    int*      win   = (int*)alloc((size_t)NQ * 4);
    _Float16* tok16 = (_Float16*)alloc((size_t)BATCH * NL * HDIM * 2);
    _Float16* xq16  = (_Float16*)alloc((size_t)NQ * HDIM * 2);
    float*    hb0   = (float*)alloc((size_t)NQ * HDIM * 4);
    float*    hb1   = (float*)alloc((size_t)NQ * HDIM * 4);
    float*    qbuf  = (float*)alloc((size_t)NQ * INNER * 4);
    _Float16* kv16  = (_Float16*)alloc((size_t)BATCH * NL * KVW * 2);
    _Float16* obuf  = (_Float16*)alloc((size_t)BQ * INNER * 2);
    _Float16* mod16 = (_Float16*)alloc((size_t)BQ * HDIM * 2);
    float*    m0b   = (float*)alloc((size_t)BQ * HDIM * 4);
    float*    m1b   = (float*)alloc((size_t)BQ * HDIM * 4);
    _Float16* sum16 = (_Float16*)alloc((size_t)BQ * HDIM * 2);
    float*    hv1b  = (float*)alloc((size_t)BQ * HDIM * 4);
    _Float16* queryWt = (_Float16*)alloc((size_t)FDIM * HDIM * 2);
    _Float16* qWt     = (_Float16*)alloc((size_t)HDIM * INNER * 2);
    _Float16* kvWt    = (_Float16*)alloc((size_t)HDIM * KVW * 2);
    _Float16* outWt   = (_Float16*)alloc((size_t)INNER * HDIM * 2);
    _Float16* bandWt0 = (_Float16*)alloc((size_t)FDIM * HDIM * 2);
    _Float16* bandWt1 = (_Float16*)alloc((size_t)FDIM * HDIM * 2);
    _Float16* modWt0  = (_Float16*)alloc((size_t)HDIM * HDIM * 2);
    _Float16* modWt1  = (_Float16*)alloc((size_t)HDIM * HDIM * 2);
    _Float16* hvWt    = (_Float16*)alloc((size_t)HDIM * HDIM * 2);

    auto tpose = [&](const float* W, _Float16* Wt, int K, int N) {
        k_transpose_f16<<<(K * N + 255) / 256, 256, 0, stream>>>(W, Wt, K, N);
    };
    auto gemm = [&](const _Float16* A, const _Float16* Wt, const float* bias,
                    const float* addend, int rowmod, float* oF, _Float16* oH,
                    int M, int N, int K, int relu) {
        dim3 g(M / 64, (N + 127) / 128);
        k_gemm_wmma<<<g, 256, 0, stream>>>(A, Wt, bias, addend, rowmod,
                                           oF, oH, M, N, K, relu);
    };

    // 1. per-query precompute (windows + fourier features)
    k_precompute<<<NQ / 256, 256, 0, stream>>>(x, gD, gH, gW, gT, g128, g32, win);
    // 2. tokens -> f16
    k_f32_to_f16<<<(BATCH * NL * HDIM + 255) / 256, 256, 0, stream>>>(
        tokens, tok16, BATCH * NL * HDIM);
    // 3. weight transposes -> f16 (N-major for WMMA B fragments)
    tpose(query_W, queryWt, FDIM, HDIM);
    tpose(q_W,     qWt,     HDIM, INNER);
    tpose(kv_W,    kvWt,    HDIM, KVW);
    tpose(out_W,   outWt,   INNER, HDIM);
    tpose(band_W,                 bandWt0, FDIM, HDIM);
    tpose(band_W + FDIM * HDIM,   bandWt1, FDIM, HDIM);
    tpose(mod_W,                  modWt0,  HDIM, HDIM);
    tpose(mod_W + HDIM * HDIM,    modWt1,  HDIM, HDIM);
    tpose(hv_W,                   hvWt,    HDIM, HDIM);

    // 4. x_q = relu(gamma128 @ query_W + b)         (4096x256, K=64)
    gemm(g128, queryWt, query_b, nullptr, 0, nullptr, xq16, NQ, HDIM, FDIM, 1);
    // 5-6. band features h_l = relu(gamma_l @ band_W[l] + b[l])
    gemm(g128, bandWt0, band_b,        nullptr, 0, hb0, nullptr, NQ, HDIM, FDIM, 1);
    gemm(g32,  bandWt1, band_b + HDIM, nullptr, 0, hb1, nullptr, NQ, HDIM, FDIM, 1);
    // 7. q = x_q @ q_W                              (4096x128, K=256)
    gemm(xq16, qWt, nullptr, nullptr, 0, qbuf, nullptr, NQ, INNER, HDIM, 0);
    // 8. kv = tokens @ kv_W                         (2048x256, K=256)
    gemm(tok16, kvWt, nullptr, nullptr, 0, nullptr, kv16, BATCH * NL, KVW, HDIM, 0);
    // 9. windowed attention -> o                    (8192x128)
    k_attn<<<BQ / 8, 256, 0, stream>>>(qbuf, kv16, win, obuf);
    // 10. modulation = o @ out_W + out_b            (8192x256, K=128)
    gemm(obuf, outWt, out_b, nullptr, 0, nullptr, mod16, BQ, HDIM, INNER, 0);
    // 11-12. m_l = relu(h_l + modulation @ mod_W[l] + mod_b[l])
    gemm(mod16, modWt0, mod_b,        hb0, NQ, m0b, nullptr, BQ, HDIM, HDIM, 1);
    gemm(mod16, modWt1, mod_b + HDIM, hb1, NQ, m1b, nullptr, BQ, HDIM, HDIM, 1);
    // 13. sum16 = f16(m0 + m1)
    k_sum_f16<<<(BQ * HDIM + 255) / 256, 256, 0, stream>>>(m0b, m1b, sum16, BQ * HDIM);
    // 14. hv1 = relu((m1 + m0) @ hv_W + hv_b)
    gemm(sum16, hvWt, hv_b, nullptr, 0, hv1b, nullptr, BQ, HDIM, HDIM, 1);
    // 15. out = m0 @ outl_W0 + hv1 @ outl_W1 + b0 + b1
    k_final<<<BQ / 8, 256, 0, stream>>>(m0b, hv1b, outl_W, outl_b, outp);
}
